// TransformerBlock_395136991402
// MI455X (gfx1250) — matrix operand order repaired and checked by reading, not compiled, not run on hardware
//
#include <hip/hip_runtime.h>
#include <hip/hip_bf16.h>
#include <math.h>

// ---------------- problem constants ----------------
#define DM    1024          // d_model
#define NH    16            // heads
#define DH    64            // head dim
#define DFF   4096          // ffn dim
#define SEQ   2048
#define BATCH 2
#define TOK   (BATCH*SEQ)   // 4096 tokens

typedef __bf16 bf16_t;
typedef __attribute__((ext_vector_type(16))) __bf16 v16bf;
typedef __attribute__((ext_vector_type(8)))  float  v8f;

union FragU { unsigned u[8]; v16bf v; };           // dword-pair view of a bf16 fragment
union FragH { unsigned short h[16]; v16bf v; };    // element view of a bf16 fragment

static __device__ __forceinline__ v8f wmma_bf16(v16bf a, v16bf b, v8f c) {
  // D = A(16x32 bf16) * B(32x16 bf16) + C(16x16 f32)
  return __builtin_amdgcn_wmma_f32_16x16x32_bf16(false, a, false, b, (short)0, c, false, false);
}

// A-matrix (16x32, 16-bit) K index for VGPR v, lane-half:  ISA 7.12.2
static __device__ __forceinline__ int a_kidx(int v, int half) {
  return ((v < 4) ? (2 * v) : (2 * (v - 4) + 16)) + (half ? 8 : 0);
}
// B-matrix (32x16, 16-bit) K index for VGPR v, lane-half (same K order as A)
static __device__ __forceinline__ int b_kidx(int v, int half) {
  return ((v < 4) ? (2 * v) : (2 * (v - 4) + 16)) + (half ? 8 : 0);
}

// ---- CDNA5 async global->LDS copy (ASYNCcnt-tracked DMA, ISA 10. / 15.18.3) ----
static __device__ __forceinline__ void async_b128(void* lds, const void* g) {
  asm volatile("global_load_async_to_lds_b128 %0, %1, off"
               :: "v"((unsigned)(size_t)lds), "v"(g)
               : "memory");
}
template <int N>
static __device__ __forceinline__ void wait_async() {
  asm volatile("s_wait_asynccnt %0" :: "i"(N) : "memory");
}

// ---------------- elementwise kernels ----------------

__global__ void cvt_f32_bf16(const float* __restrict__ in, bf16_t* __restrict__ out, int n) {
  int i = blockIdx.x * blockDim.x + threadIdx.x;
  if (i < n) out[i] = (bf16_t)in[i];
}

// one block (256 threads) per token row of DM=1024
__global__ __launch_bounds__(256)
void rmsnorm_kernel(const float* __restrict__ X, const float* __restrict__ G,
                    bf16_t* __restrict__ Out) {
  __shared__ float red[256];
  const int t = blockIdx.x;
  const int tid = threadIdx.x;
  const float4 xv = *(const float4*)(X + (size_t)t * DM + tid * 4);
  red[tid] = xv.x * xv.x + xv.y * xv.y + xv.z * xv.z + xv.w * xv.w;
  __syncthreads();
  for (int s = 128; s > 0; s >>= 1) {
    if (tid < s) red[tid] += red[tid + s];
    __syncthreads();
  }
  const float rinv = rsqrtf(red[0] * (1.0f / DM) + 1e-5f);
  const float4 gv = *(const float4*)(G + tid * 4);
  const size_t o = (size_t)t * DM + tid * 4;
  Out[o + 0] = (bf16_t)(xv.x * rinv * gv.x);
  Out[o + 1] = (bf16_t)(xv.y * rinv * gv.y);
  Out[o + 2] = (bf16_t)(xv.z * rinv * gv.z);
  Out[o + 3] = (bf16_t)(xv.w * rinv * gv.w);
}

// in-place RoPE on bf16 Q and K: one thread per (token, head, pair)
__global__ void rope_kernel(bf16_t* __restrict__ Qm, bf16_t* __restrict__ Km) {
  int id = blockIdx.x * blockDim.x + threadIdx.x;
  if (id >= TOK * NH * (DH / 2)) return;
  const int i = id & 31;              // pair index 0..31
  const int rest = id >> 5;
  const int h = rest & (NH - 1);
  const int t = rest >> 4;            // token 0..TOK-1
  const int s = t & (SEQ - 1);        // position within sequence
  const float invf = __powf(10000.0f, -(float)i / 32.0f);  // theta^(-2i/64)
  float sn, cs;
  sincosf((float)s * invf, &sn, &cs);
  const size_t off = (size_t)t * DM + h * DH + 2 * i;
  const float q1 = (float)Qm[off], q2 = (float)Qm[off + 1];
  Qm[off]     = (bf16_t)(q1 * cs - q2 * sn);
  Qm[off + 1] = (bf16_t)(q1 * sn + q2 * cs);
  const float k1 = (float)Km[off], k2 = (float)Km[off + 1];
  Km[off]     = (bf16_t)(k1 * cs - k2 * sn);
  Km[off + 1] = (bf16_t)(k1 * sn + k2 * cs);
}

// silu(u1) * u3 -> bf16
__global__ void gate_kernel(const bf16_t* __restrict__ u1, const bf16_t* __restrict__ u3,
                            bf16_t* __restrict__ g, int n) {
  int i = blockIdx.x * blockDim.x + threadIdx.x;
  if (i >= n) return;
  const float a = (float)u1[i];
  const float b = (float)u3[i];
  g[i] = (bf16_t)(a / (1.0f + __expf(-a)) * b);
}

// ---------------- WMMA GEMM:  C[M,N] = A[M,K](bf16) @ W[N,K]^T(bf16) ----------------
// block tile 128x128, 8 waves (4 along M x 2 along N), wave tile 32x64.
// K-step 64 (16 WMMAs per wave per barrier pair), double-buffered async global->LDS staging.
// EPI==0: store bf16.  EPI==1: store f32 with residual add.
template <int EPI>
__global__ __launch_bounds__(256)
void gemm_wmma(const bf16_t* __restrict__ A, const bf16_t* __restrict__ Bw,
               void* __restrict__ outp, const float* __restrict__ resid,
               int M, int N, int K) {
  __shared__ __align__(16) bf16_t As[2][128][72];  // 64 + 8 pad
  __shared__ __align__(16) bf16_t Bs[2][128][72];

  const int tid  = threadIdx.x;
  const int lane = tid & 31;
  const int wave = tid >> 5;
  const int half = lane >> 4;
  const int l15  = lane & 15;
  const int wm = wave & 3;       // 4 waves along M (32 rows each)
  const int wn = wave >> 2;      // 2 waves along N (64 cols each)
  const int m0 = blockIdx.y * 128;
  const int n0 = blockIdx.x * 128;

  v8f acc[2][4];
#pragma unroll
  for (int i = 0; i < 2; i++)
#pragma unroll
    for (int j = 0; j < 4; j++) acc[i][j] = (v8f){0, 0, 0, 0, 0, 0, 0, 0};

  // staging map: 128 rows x 64 cols, 32 elems (= 4x b128) per thread, for A and B alike
  const int srow = tid >> 1, scol = (tid & 1) * 32;
  const bf16_t* gA = A  + (size_t)(m0 + srow) * K + scol;
  const bf16_t* gB = Bw + (size_t)(n0 + srow) * K + scol;

  auto stage = [&](int buf, int k0) {
#pragma unroll
    for (int c = 0; c < 4; c++)
      async_b128(&As[buf][srow][scol + c * 8], gA + k0 + c * 8);
#pragma unroll
    for (int c = 0; c < 4; c++)
      async_b128(&Bs[buf][srow][scol + c * 8], gB + k0 + c * 8);
  };

  const int nt = K / 64;
  stage(0, 0);

  for (int it = 0; it < nt; ++it) {
    const int cur = it & 1;
    if (it + 1 < nt) {
      stage(1 - cur, (it + 1) * 64);   // overlap next tile DMA with current compute
      wait_async<8>();                 // current tile's 8 copies complete (in-order)
    } else {
      wait_async<0>();
    }
    __syncthreads();                   // current tile visible to all waves

#pragma unroll
    for (int kc = 0; kc < 64; kc += 32) {
      FragU af[2], bfr[4];
#pragma unroll
      for (int i = 0; i < 2; i++) {
        const int r = wm * 32 + i * 16 + l15;
#pragma unroll
        for (int v = 0; v < 8; v++)
          af[i].u[v] = *(const unsigned*)(&As[cur][r][kc + a_kidx(v, half)]);
      }
#pragma unroll
      for (int j = 0; j < 4; j++) {
        const int r = wn * 64 + j * 16 + l15;
#pragma unroll
        for (int v = 0; v < 8; v++)
          bfr[j].u[v] = *(const unsigned*)(&Bs[cur][r][kc + b_kidx(v, half)]);
      }
#pragma unroll
      for (int i = 0; i < 2; i++)
#pragma unroll
        for (int j = 0; j < 4; j++)
          acc[i][j] = wmma_bf16(af[i].v, bfr[j].v, acc[i][j]);
    }

    __syncthreads();                   // all waves done reading 'cur' before it is re-staged
  }

  // C layout: VGPR r holds row (r + 8*half), column = lane&15  (ISA 7.12.2)
#pragma unroll
  for (int i = 0; i < 2; i++)
#pragma unroll
    for (int j = 0; j < 4; j++)
#pragma unroll
      for (int r = 0; r < 8; r++) {
        const int gm = m0 + wm * 32 + i * 16 + r + 8 * half;
        const int gn = n0 + wn * 64 + j * 16 + l15;
        const float val = acc[i][j][r];
        const size_t off = (size_t)gm * N + gn;
        if (EPI == 0) ((bf16_t*)outp)[off] = (bf16_t)val;
        else          ((float*)outp)[off] = val + resid[off];
      }
}

// ---------------- flash attention (causal), WMMA for QK^T and P*V ----------------
// 4 waves / block; each wave owns TWO 16-row q tiles (32 query rows) so K-fragments and the
// V LDS tile are reused across both. Q/K/V layout: [B*S][DM], head at col h*64.
__global__ __launch_bounds__(128)
void attn_wmma(const bf16_t* __restrict__ Qm, const bf16_t* __restrict__ Km,
               const bf16_t* __restrict__ Vm, bf16_t* __restrict__ Om) {
  __shared__ __align__(16) bf16_t Pt[4][2][16][32];  // per-wave P tiles (C->A relayout)
  __shared__ __align__(16) bf16_t Vs[4][32][64];     // per-wave V chunk (32 keys x 64 ch)

  const int tid  = threadIdx.x;
  const int lane = tid & 31;
  const int w    = tid >> 5;
  const int half = lane >> 4;
  const int l15  = lane & 15;
  const int q0 = (blockIdx.x * 4 + w) * 32;   // 32 query rows per wave
  const int h  = blockIdx.y;
  const int b  = blockIdx.z;
  const size_t base = (size_t)b * SEQ * DM + (size_t)h * DH;

  // Q fragments (A layout) for both 16-row tiles, channel chunks [0,32) and [32,64)
  FragU qa[2][2];
#pragma unroll
  for (int t = 0; t < 2; t++) {
    const size_t qrow = base + (size_t)(q0 + 16 * t + l15) * DM;
#pragma unroll
    for (int cc = 0; cc < 2; cc++)
#pragma unroll
      for (int v = 0; v < 8; v++)
        qa[t][cc].u[v] = *(const unsigned*)(Qm + qrow + cc * 32 + a_kidx(v, half));
  }

  float m_st[2][8], l_st[2][8];
  v8f oacc[2][4];
#pragma unroll
  for (int t = 0; t < 2; t++) {
#pragma unroll
    for (int r = 0; r < 8; r++) { m_st[t][r] = -3e38f; l_st[t][r] = 0.0f; }
#pragma unroll
    for (int cg = 0; cg < 4; cg++) oacc[t][cg] = (v8f){0, 0, 0, 0, 0, 0, 0, 0};
  }

  for (int kb0 = 0; kb0 <= q0 + 31; kb0 += 32) {
    // async-stage V chunk: lane DMA-copies one key row (wave-private, ASYNCcnt tracked)
    {
      const int key = kb0 + lane;                  // <= q0+31 <= SEQ-1 by construction
      const bf16_t* gv = Vm + base + (size_t)key * DM;
#pragma unroll
      for (int t = 0; t < 8; t++)
        async_b128(&Vs[w][lane][t * 8], gv + t * 8);
      // prefetch next chunk's K row into cache while we work on this one
      const int keyn = (key + 32) < SEQ ? (key + 32) : SEQ - 1;
      __builtin_prefetch(Km + base + (size_t)keyn * DM, 0, 1);
    }

    float st[2][2][8];                             // [qtile][keytile][r]
    float tm[2][8];
#pragma unroll
    for (int t = 0; t < 2; t++)
#pragma unroll
      for (int r = 0; r < 8; r++) tm[t][r] = -3e38f;

#pragma unroll
    for (int j = 0; j < 2; j++) {
      const int key = kb0 + 16 * j + l15;          // this lane's key column
      const size_t krow = base + (size_t)key * DM;
      FragU kf[2];                                 // B layout: K dim = channel, N = key
#pragma unroll
      for (int cc = 0; cc < 2; cc++)
#pragma unroll
        for (int v = 0; v < 8; v++)
          kf[cc].u[v] = *(const unsigned*)(Km + krow + cc * 32 + b_kidx(v, half));
#pragma unroll
      for (int t = 0; t < 2; t++) {                // reuse kf across both q tiles
        v8f sacc = (v8f){0, 0, 0, 0, 0, 0, 0, 0};
        sacc = wmma_bf16(qa[t][0].v, kf[0].v, sacc);
        sacc = wmma_bf16(qa[t][1].v, kf[1].v, sacc);
#pragma unroll
        for (int r = 0; r < 8; r++) {
          float sv = sacc[r] * 0.125f;             // 1/sqrt(64)
          const int qrow = q0 + 16 * t + r + 8 * half;
          if (key > qrow) sv = -3e38f;             // causal mask
          st[t][j][r] = sv;
          tm[t][r] = fmaxf(tm[t][r], sv);
        }
      }
    }

    FragU pa[2];
#pragma unroll
    for (int t = 0; t < 2; t++) {
      // row max across the 16 lanes holding each row
#pragma unroll
      for (int r = 0; r < 8; r++) {
        float v = tm[t][r];
        v = fmaxf(v, __shfl_xor(v, 1, 16));
        v = fmaxf(v, __shfl_xor(v, 2, 16));
        v = fmaxf(v, __shfl_xor(v, 4, 16));
        v = fmaxf(v, __shfl_xor(v, 8, 16));
        tm[t][r] = v;
      }
      float alpha[8];
#pragma unroll
      for (int r = 0; r < 8; r++) {
        const float nm = fmaxf(m_st[t][r], tm[t][r]);
        alpha[r] = __expf(m_st[t][r] - nm);
        m_st[t][r] = nm;
      }
#pragma unroll
      for (int j = 0; j < 2; j++)
#pragma unroll
        for (int r = 0; r < 8; r++) st[t][j][r] = __expf(st[t][j][r] - m_st[t][r]);
#pragma unroll
      for (int r = 0; r < 8; r++) {
        float v = st[t][0][r] + st[t][1][r];
        v += __shfl_xor(v, 1, 16);
        v += __shfl_xor(v, 2, 16);
        v += __shfl_xor(v, 4, 16);
        v += __shfl_xor(v, 8, 16);
        l_st[t][r] = l_st[t][r] * alpha[r] + v;
      }
#pragma unroll
      for (int cg = 0; cg < 4; cg++)
#pragma unroll
        for (int r = 0; r < 8; r++) oacc[t][cg][r] *= alpha[r];

      // relayout P: C layout -> LDS -> A layout (LDS ops in-order within a wave)
#pragma unroll
      for (int j = 0; j < 2; j++)
#pragma unroll
        for (int r = 0; r < 8; r++)
          Pt[w][t][r + 8 * half][16 * j + l15] = (bf16_t)st[t][j][r];
#pragma unroll
      for (int v = 0; v < 8; v++)
        pa[t].u[v] = *(const unsigned*)(&Pt[w][t][l15][a_kidx(v, half)]);
    }

    wait_async<0>();   // V chunk landed in LDS (wave-private -> no barrier needed)

#pragma unroll
    for (int cg = 0; cg < 4; cg++) {
      FragH vf;   // B layout: K = key (rows of Vs), N = channel; reused across both q tiles
#pragma unroll
      for (int v = 0; v < 8; v++) {
        const int k0 = b_kidx(v, half);
        vf.h[2 * v]     = *(const unsigned short*)(&Vs[w][k0][cg * 16 + l15]);
        vf.h[2 * v + 1] = *(const unsigned short*)(&Vs[w][k0 + 1][cg * 16 + l15]);
      }
#pragma unroll
      for (int t = 0; t < 2; t++)
        oacc[t][cg] = wmma_bf16(pa[t].v, vf.v, oacc[t][cg]);
    }
  }

  // normalize and store bf16
#pragma unroll
  for (int t = 0; t < 2; t++)
#pragma unroll
    for (int cg = 0; cg < 4; cg++)
#pragma unroll
      for (int r = 0; r < 8; r++) {
        const int tok = q0 + 16 * t + r + 8 * half;
        const float val = oacc[t][cg][r] / l_st[t][r];
        Om[base + (size_t)tok * DM + cg * 16 + l15] = (bf16_t)val;
      }
}

// ---------------- launch ----------------
extern "C" void kernel_launch(void* const* d_in, const int* in_sizes, int n_in,
                              void* d_out, int out_size, void* d_ws, size_t ws_size,
                              hipStream_t stream) {
  const float* x  = (const float*)d_in[0];
  const float* WQ = (const float*)d_in[1];
  const float* WK = (const float*)d_in[2];
  const float* WV = (const float*)d_in[3];
  const float* WO = (const float*)d_in[4];
  const float* W1 = (const float*)d_in[5];
  const float* W2 = (const float*)d_in[6];
  const float* W3 = (const float*)d_in[7];
  const float* G1 = (const float*)d_in[8];
  const float* G2 = (const float*)d_in[9];

  char* ws = (char*)d_ws;
  size_t off = 0;
  auto alloc = [&](size_t bytes) -> void* {
    void* p = ws + off;
    off = (off + bytes + 255) & ~(size_t)255;
    return p;
  };

  bf16_t* wqb = (bf16_t*)alloc((size_t)DM * DM * 2);
  bf16_t* wkb = (bf16_t*)alloc((size_t)DM * DM * 2);
  bf16_t* wvb = (bf16_t*)alloc((size_t)DM * DM * 2);
  bf16_t* wob = (bf16_t*)alloc((size_t)DM * DM * 2);
  bf16_t* w1b = (bf16_t*)alloc((size_t)DFF * DM * 2);
  bf16_t* w3b = (bf16_t*)alloc((size_t)DFF * DM * 2);
  bf16_t* w2b = (bf16_t*)alloc((size_t)DM * DFF * 2);
  bf16_t* hb  = (bf16_t*)alloc((size_t)TOK * DM * 2);
  bf16_t* qb  = (bf16_t*)alloc((size_t)TOK * DM * 2);
  bf16_t* kb  = (bf16_t*)alloc((size_t)TOK * DM * 2);
  bf16_t* vb  = (bf16_t*)alloc((size_t)TOK * DM * 2);
  bf16_t* ao  = (bf16_t*)alloc((size_t)TOK * DM * 2);
  float*  x1  = (float*)alloc((size_t)TOK * DM * 4);
  bf16_t* h2b = (bf16_t*)alloc((size_t)TOK * DM * 2);
  bf16_t* u1  = (bf16_t*)alloc((size_t)TOK * DFF * 2);
  bf16_t* u3  = (bf16_t*)alloc((size_t)TOK * DFF * 2);
  bf16_t* gb  = (bf16_t*)alloc((size_t)TOK * DFF * 2);

  const int nA = DM * DM, nF = DFF * DM;
  cvt_f32_bf16<<<(nA + 255) / 256, 256, 0, stream>>>(WQ, wqb, nA);
  cvt_f32_bf16<<<(nA + 255) / 256, 256, 0, stream>>>(WK, wkb, nA);
  cvt_f32_bf16<<<(nA + 255) / 256, 256, 0, stream>>>(WV, wvb, nA);
  cvt_f32_bf16<<<(nA + 255) / 256, 256, 0, stream>>>(WO, wob, nA);
  cvt_f32_bf16<<<(nF + 255) / 256, 256, 0, stream>>>(W1, w1b, nF);
  cvt_f32_bf16<<<(nF + 255) / 256, 256, 0, stream>>>(W3, w3b, nF);
  cvt_f32_bf16<<<(nF + 255) / 256, 256, 0, stream>>>(W2, w2b, nF);

  rmsnorm_kernel<<<TOK, 256, 0, stream>>>(x, G1, hb);

  dim3 gp(DM / 128, TOK / 128);
  gemm_wmma<0><<<gp, 256, 0, stream>>>(hb, wqb, (void*)qb, nullptr, TOK, DM, DM);
  gemm_wmma<0><<<gp, 256, 0, stream>>>(hb, wkb, (void*)kb, nullptr, TOK, DM, DM);
  gemm_wmma<0><<<gp, 256, 0, stream>>>(hb, wvb, (void*)vb, nullptr, TOK, DM, DM);

  const int nr = TOK * NH * (DH / 2);
  rope_kernel<<<(nr + 255) / 256, 256, 0, stream>>>(qb, kb);

  attn_wmma<<<dim3(SEQ / 128, NH, BATCH), 128, 0, stream>>>(qb, kb, vb, ao);

  gemm_wmma<1><<<gp, 256, 0, stream>>>(ao, wob, (void*)x1, x, TOK, DM, DM);

  rmsnorm_kernel<<<TOK, 256, 0, stream>>>(x1, G2, h2b);

  dim3 gf(DFF / 128, TOK / 128);
  gemm_wmma<0><<<gf, 256, 0, stream>>>(h2b, w1b, (void*)u1, nullptr, TOK, DFF, DM);
  gemm_wmma<0><<<gf, 256, 0, stream>>>(h2b, w3b, (void*)u3, nullptr, TOK, DFF, DM);

  const int ng = TOK * DFF;
  gate_kernel<<<(ng + 255) / 256, 256, 0, stream>>>(u1, u3, gb, ng);

  gemm_wmma<1><<<gp, 256, 0, stream>>>(gb, w2b, d_out, x1, TOK, DM, DFF);
}
